// CMultiHeadAttention_40973988004751
// MI455X (gfx1250) — matrix operand order repaired and checked by reading, not compiled, not run on hardware
//
#include <hip/hip_runtime.h>
#include <hip/hip_bf16.h>

typedef _Float16 half_t;
typedef __attribute__((ext_vector_type(16))) _Float16 v16h;
typedef __attribute__((ext_vector_type(4)))  _Float16 v4h;
typedef __attribute__((ext_vector_type(8)))  float    v8f;

#define B_  4
#define L_  2048
#define D_  512
#define H_  8
#define DH_ 64
#define BL_ (B_*L_)   // 8192
#define BH_ (B_*H_)   // 32
#define TM  128       // M workgroup tile (2 m-subtiles per wave)
#define TS  64        // N workgroup tile
#define KS  32        // K step (wmma f16 k=32)
#define LDT 40        // padded LDS row stride in halfs (80B: 16B-aligned, bank-spread)

// ---------------- CDNA5 async global->LDS copy (ASYNCcnt path) ----------------
__device__ __forceinline__ void async_ld16(void* lds_ptr, const void* gptr) {
  const unsigned int lds_off = (unsigned int)(size_t)lds_ptr;   // low 32 bits = LDS offset
  asm volatile("global_load_async_to_lds_b128 %0, %1, off"
               :: "v"(lds_off), "v"(gptr) : "memory");
}
__device__ __forceinline__ void wait_async() {
  asm volatile("s_wait_asynccnt 0x0" ::: "memory");
}

// ---------------- WMMA fragment helpers (wave32, 16x16x32 f16) ----------------
__device__ __forceinline__ int lane_id() { return (int)(threadIdx.x & 31u); }

// A fragment: A[m][k], m in 0..15, k in 0..31, src row-major with leading dim ld.
__device__ __forceinline__ v16h frag_a_load(const half_t* __restrict__ src, int ld) {
  const int lane = lane_id();
  const int m  = lane & 15;
  const int kb = (lane >> 4) << 4;           // 0 or 16
  v16h a;
#pragma unroll
  for (int j = 0; j < 8; ++j) {
    const int kk = kb + 2*j;  // kb+2*j
    a[2*j]   = src[m * ld + kk];
    a[2*j+1] = src[m * ld + kk + 1];
  }
  return a;
}

// B fragment where B[k][n] = S[n][k] (B = S^T, S row-major, ld = its stride).
__device__ __forceinline__ v16h frag_b_from_rows(const half_t* __restrict__ src, int ld) {
  const int lane = lane_id();
  const int n  = lane & 15;
  const int kb = (lane >> 4) << 4;           // 0 or 16
  v16h b;
#pragma unroll
  for (int j = 0; j < 8; ++j) {
    b[2*j]   = src[n * ld + kb + 2*j];
    b[2*j+1] = src[n * ld + kb + 2*j + 1];
  }
  return b;
}

// B fragment where B[k][n] = S[k][n] (S row-major [K][N], ld = its stride).
__device__ __forceinline__ v16h frag_b_load(const half_t* __restrict__ src, int ld) {
  const int lane = lane_id();
  const int n  = lane & 15;
  const int kb = (lane >> 4) << 4;
  v16h b;
#pragma unroll
  for (int j = 0; j < 8; ++j) {
    b[2*j]   = src[(kb + 2*j    ) * ld + n];
    b[2*j+1] = src[(kb + 2*j + 1) * ld + n];
  }
  return b;
}

__device__ __forceinline__ v8f wmma_f16(v16h a, v16h b, v8f c) {
  return __builtin_amdgcn_wmma_f32_16x16x32_f16(false, a, false, b, (short)0, c,
                                                false, false);
}

__device__ __forceinline__ void zero8(v8f& v) {
#pragma unroll
  for (int e = 0; e < 8; ++e) v[e] = 0.0f;
}

// ---------------- 0: f32 -> f16 pack (vectorized x4) ----------------
__global__ void cvt_f32_f16(const float* __restrict__ src, half_t* __restrict__ dst,
                            int n4) {
  int i = blockIdx.x * blockDim.x + threadIdx.x;
  if (i < n4) {
    const float4 f = ((const float4*)src)[i];
    v4h h;
    h[0] = (half_t)f.x; h[1] = (half_t)f.y; h[2] = (half_t)f.z; h[3] = (half_t)f.w;
    ((v4h*)dst)[i] = h;
  }
}

// ---------------- 1: projection GEMM  Y = X @ W^T + b  (f16 in, f16 head-split out)
// 128x64 tile / WG, double-buffered async LDS staging, 8 WMMAs per wave per K-step.
__global__ __launch_bounds__(128)
void proj_gemm(const half_t* __restrict__ X, const half_t* __restrict__ W,
               const float* __restrict__ bias, half_t* __restrict__ Yp) {
  __shared__ half_t As[2][TM * LDT];
  __shared__ half_t Bs[2][TS * LDT];
  const int tid  = threadIdx.x;
  const int wave = tid >> 5;
  const int n0 = blockIdx.x * TS;
  const int m0 = blockIdx.y * TM;

  auto stage = [&](int p, int k0) {
    // A tile: 128 rows x 32 halfs = 512 x 16B chunks -> 4 per thread (uniform)
#pragma unroll
    for (int it = 0; it < 4; ++it) {
      const int c = tid + it * 128;
      const int r = c >> 2, cc = c & 3;
      async_ld16(&As[p][r * LDT + cc * 8],
                 &X[(size_t)(m0 + r) * D_ + k0 + cc * 8]);
    }
    // B tile (weight rows): 256 chunks -> 2 per thread (uniform)
#pragma unroll
    for (int it = 0; it < 2; ++it) {
      const int c = tid + it * 128;
      const int r = c >> 2, cc = c & 3;
      async_ld16(&Bs[p][r * LDT + cc * 8],
                 &W[(size_t)(n0 + r) * D_ + k0 + cc * 8]);
    }
  };

  v8f acc[2][4];
#pragma unroll
  for (int mi = 0; mi < 2; ++mi)
#pragma unroll
    for (int ni = 0; ni < 4; ++ni) zero8(acc[mi][ni]);

  stage(0, 0);
  wait_async();
  __syncthreads();

  int p = 0;
  for (int k0 = 0; k0 < D_; k0 += KS) {
    if (k0 + KS < D_) stage(p ^ 1, k0 + KS);   // DMA next tile under the WMMAs
    v16h a[2];
#pragma unroll
    for (int mi = 0; mi < 2; ++mi)
      a[mi] = frag_a_load(&As[p][((wave * 2 + mi) * 16) * LDT], LDT);
#pragma unroll
    for (int ni = 0; ni < 4; ++ni) {
      const v16h bfr = frag_b_from_rows(&Bs[p][(ni * 16) * LDT], LDT);
#pragma unroll
      for (int mi = 0; mi < 2; ++mi)
        acc[mi][ni] = wmma_f16(a[mi], bfr, acc[mi][ni]);
    }
    wait_async();
    __syncthreads();
    p ^= 1;
  }

  // epilogue: C/D layout -> lane%16 = n, rows (lane/16)*8 + r
  const int lane = lane_id();
  const int nloc = lane & 15;
  const int mb   = (lane >> 4) << 3;
#pragma unroll
  for (int mi = 0; mi < 2; ++mi) {
#pragma unroll
    for (int ni = 0; ni < 4; ++ni) {
      const int n = n0 + ni * 16 + nloc;
      const int h = n >> 6, dd = n & (DH_ - 1);
      const float bv = bias[n];
#pragma unroll
      for (int r = 0; r < 8; ++r) {
        const int m = m0 + (wave * 2 + mi) * 16 + mb + r;
        const int bb = m >> 11, l = m & (L_ - 1);
        Yp[(size_t)((bb * H_ + h) * L_ + l) * DH_ + dd] = (half_t)(acc[mi][ni][r] + bv);
      }
    }
  }
}

// ---------------- 2: scores  S = Qp Kp^T * 0.125 + mask*(-1e9)  (f32 -> attn, NT)
__global__ __launch_bounds__(128)
void scores_kernel(const half_t* __restrict__ Qp, const half_t* __restrict__ Kp,
                   const int* __restrict__ mask, float* __restrict__ attn) {
  const int wave = threadIdx.x >> 5;
  const int n0 = blockIdx.x * TS;
  const int m0 = blockIdx.y * TM;
  const int bh = blockIdx.z;
  const int b  = bh >> 3;
  const half_t* Qb = Qp + (size_t)bh * L_ * DH_;
  const half_t* Kb = Kp + (size_t)bh * L_ * DH_;

  v8f acc[2][4];
#pragma unroll
  for (int mi = 0; mi < 2; ++mi)
#pragma unroll
    for (int ni = 0; ni < 4; ++ni) zero8(acc[mi][ni]);

#pragma unroll
  for (int k0 = 0; k0 < DH_; k0 += KS) {
    v16h a[2];
#pragma unroll
    for (int mi = 0; mi < 2; ++mi)
      a[mi] = frag_a_load(Qb + (size_t)(m0 + (wave * 2 + mi) * 16) * DH_ + k0, DH_);
#pragma unroll
    for (int ni = 0; ni < 4; ++ni) {
      const v16h bfr = frag_b_from_rows(Kb + (size_t)(n0 + ni * 16) * DH_ + k0, DH_);
#pragma unroll
      for (int mi = 0; mi < 2; ++mi)
        acc[mi][ni] = wmma_f16(a[mi], bfr, acc[mi][ni]);
    }
  }
  const int lane = lane_id();
  const int nloc = lane & 15;
  const int mb   = (lane >> 4) << 3;
  float* Ab = attn + (size_t)bh * L_ * L_;
#pragma unroll
  for (int mi = 0; mi < 2; ++mi) {
#pragma unroll
    for (int ni = 0; ni < 4; ++ni) {
      const int n = n0 + ni * 16 + nloc;
      const float madd = (float)mask[b * L_ + n] * -1e9f;
#pragma unroll
      for (int r = 0; r < 8; ++r) {
        const int m = m0 + (wave * 2 + mi) * 16 + mb + r;
        __builtin_nontemporal_store(acc[mi][ni][r] * 0.125f + madd,
                                    &Ab[(size_t)m * L_ + n]);
      }
    }
  }
}

// ---------------- 3: row softmax over attn (row length L_=2048, block=256)
__global__ __launch_bounds__(256)
void softmax_rows(float* __restrict__ attn) {
  float* p = attn + (size_t)blockIdx.x * L_;
  const int tid = threadIdx.x;
  float v[8];
  float lmax = -3.4e38f;
#pragma unroll
  for (int j = 0; j < 8; ++j) { v[j] = p[tid + j * 256]; lmax = fmaxf(lmax, v[j]); }
#pragma unroll
  for (int off = 16; off; off >>= 1) lmax = fmaxf(lmax, __shfl_xor(lmax, off, 32));
  __shared__ float redm[8];
  __shared__ float reds[8];
  if ((tid & 31) == 0) redm[tid >> 5] = lmax;
  __syncthreads();
  float m = redm[0];
#pragma unroll
  for (int w = 1; w < 8; ++w) m = fmaxf(m, redm[w]);
  float lsum = 0.0f;
#pragma unroll
  for (int j = 0; j < 8; ++j) { v[j] = __expf(v[j] - m); lsum += v[j]; }
#pragma unroll
  for (int off = 16; off; off >>= 1) lsum += __shfl_xor(lsum, off, 32);
  if ((tid & 31) == 0) reds[tid >> 5] = lsum;
  __syncthreads();
  float s = 0.0f;
#pragma unroll
  for (int w = 0; w < 8; ++w) s += reds[w];
  const float inv = 1.0f / s;
#pragma unroll
  for (int j = 0; j < 8; ++j) p[tid + j * 256] = v[j] * inv;
}

// ---------------- 4: O = attn @ Vp  (f32 probs -> f16 frags, f16 concat-head out)
__global__ __launch_bounds__(128)
void attnv_gemm(const float* __restrict__ attn, const half_t* __restrict__ Vp,
                half_t* __restrict__ Oh) {
  __shared__ half_t Ps[2][TM * LDT];   // probs tile  [128][32] f16 (padded)
  __shared__ half_t Vs[2][KS * DH_];   // V tile      [32][64] f16
  const int tid  = threadIdx.x;
  const int wave = tid >> 5;
  const int bh = blockIdx.x;
  const int m0 = blockIdx.y * TM;
  const float*  Pb = attn + (size_t)bh * L_ * L_ + (size_t)m0 * L_;
  const half_t* Vb = Vp   + (size_t)bh * L_ * DH_;

  auto stage = [&](int p, int k0) {
    // async stage V tile rows k0..k0+31: 128 x 16B chunks -> exactly 1 per thread
    {
      const int r = tid >> 2, cc = tid & 3;
      async_ld16(&Vs[p][r * DH_ + cc * 8],
                 &Vb[(size_t)(k0 + r) * DH_ + cc * 8]);
    }
    // convert-stage probs: 128 rows x 32 f32 -> f16 (float4, 8 per thread, uniform)
#pragma unroll
    for (int it = 0; it < 8; ++it) {
      const int idx = tid + it * 128;
      const int r = idx >> 3, c4 = idx & 7;
      const float4 f = *(const float4*)&Pb[(size_t)r * L_ + k0 + c4 * 4];
      v4h h;
      h[0] = (half_t)f.x; h[1] = (half_t)f.y; h[2] = (half_t)f.z; h[3] = (half_t)f.w;
      *(v4h*)&Ps[p][r * LDT + c4 * 4] = h;
    }
  };

  v8f acc[2][4];
#pragma unroll
  for (int mi = 0; mi < 2; ++mi)
#pragma unroll
    for (int ni = 0; ni < 4; ++ni) zero8(acc[mi][ni]);

  stage(0, 0);
  wait_async();
  __syncthreads();

  int p = 0;
  for (int k0 = 0; k0 < L_; k0 += KS) {
    if (k0 + KS < L_) stage(p ^ 1, k0 + KS);
    v16h a[2];
#pragma unroll
    for (int mi = 0; mi < 2; ++mi)
      a[mi] = frag_a_load(&Ps[p][((wave * 2 + mi) * 16) * LDT], LDT);
#pragma unroll
    for (int ni = 0; ni < 4; ++ni) {
      const v16h bfr = frag_b_load(&Vs[p][ni * 16], DH_);
#pragma unroll
      for (int mi = 0; mi < 2; ++mi)
        acc[mi][ni] = wmma_f16(a[mi], bfr, acc[mi][ni]);
    }
    wait_async();
    __syncthreads();
    p ^= 1;
  }

  const int lane = lane_id();
  const int nloc = lane & 15;
  const int mb   = (lane >> 4) << 3;
  const int b = bh >> 3, h = bh & 7;
#pragma unroll
  for (int mi = 0; mi < 2; ++mi) {
#pragma unroll
    for (int ni = 0; ni < 4; ++ni) {
      const int dd = ni * 16 + nloc;
#pragma unroll
      for (int r = 0; r < 8; ++r) {
        const int l = m0 + (wave * 2 + mi) * 16 + mb + r;
        Oh[(size_t)(b * L_ + l) * D_ + h * DH_ + dd] = (half_t)acc[mi][ni][r];
      }
    }
  }
}

// ---------------- 5: output projection  out = Oh @ Wo^T + bo  (f32 out)
__global__ __launch_bounds__(128)
void out_proj_gemm(const half_t* __restrict__ X, const half_t* __restrict__ W,
                   const float* __restrict__ bias, float* __restrict__ Y) {
  __shared__ half_t As[2][TM * LDT];
  __shared__ half_t Bs[2][TS * LDT];
  const int tid  = threadIdx.x;
  const int wave = tid >> 5;
  const int n0 = blockIdx.x * TS;
  const int m0 = blockIdx.y * TM;

  auto stage = [&](int p, int k0) {
#pragma unroll
    for (int it = 0; it < 4; ++it) {
      const int c = tid + it * 128;
      const int r = c >> 2, cc = c & 3;
      async_ld16(&As[p][r * LDT + cc * 8],
                 &X[(size_t)(m0 + r) * D_ + k0 + cc * 8]);
    }
#pragma unroll
    for (int it = 0; it < 2; ++it) {
      const int c = tid + it * 128;
      const int r = c >> 2, cc = c & 3;
      async_ld16(&Bs[p][r * LDT + cc * 8],
                 &W[(size_t)(n0 + r) * D_ + k0 + cc * 8]);
    }
  };

  v8f acc[2][4];
#pragma unroll
  for (int mi = 0; mi < 2; ++mi)
#pragma unroll
    for (int ni = 0; ni < 4; ++ni) zero8(acc[mi][ni]);

  stage(0, 0);
  wait_async();
  __syncthreads();

  int p = 0;
  for (int k0 = 0; k0 < D_; k0 += KS) {
    if (k0 + KS < D_) stage(p ^ 1, k0 + KS);
    v16h a[2];
#pragma unroll
    for (int mi = 0; mi < 2; ++mi)
      a[mi] = frag_a_load(&As[p][((wave * 2 + mi) * 16) * LDT], LDT);
#pragma unroll
    for (int ni = 0; ni < 4; ++ni) {
      const v16h bfr = frag_b_from_rows(&Bs[p][(ni * 16) * LDT], LDT);
#pragma unroll
      for (int mi = 0; mi < 2; ++mi)
        acc[mi][ni] = wmma_f16(a[mi], bfr, acc[mi][ni]);
    }
    wait_async();
    __syncthreads();
    p ^= 1;
  }

  const int lane = lane_id();
  const int nloc = lane & 15;
  const int mb   = (lane >> 4) << 3;
#pragma unroll
  for (int mi = 0; mi < 2; ++mi) {
#pragma unroll
    for (int ni = 0; ni < 4; ++ni) {
      const int n = n0 + ni * 16 + nloc;
      const float bv = bias[n];
#pragma unroll
      for (int r = 0; r < 8; ++r) {
        const int m = m0 + (wave * 2 + mi) * 16 + mb + r;
        Y[(size_t)m * D_ + n] = acc[mi][ni][r] + bv;
      }
    }
  }
}

// ---------------- host launcher ----------------
extern "C" void kernel_launch(void* const* d_in, const int* in_sizes, int n_in,
                              void* d_out, int out_size, void* d_ws, size_t ws_size,
                              hipStream_t stream) {
  (void)in_sizes; (void)n_in; (void)out_size; (void)ws_size;
  const float* q    = (const float*)d_in[0];
  const float* k    = (const float*)d_in[1];
  const float* v    = (const float*)d_in[2];
  const int*   mask = (const int*)  d_in[3];
  const float* wq_w = (const float*)d_in[4];
  const float* wq_b = (const float*)d_in[5];
  const float* wk_w = (const float*)d_in[6];
  const float* wk_b = (const float*)d_in[7];
  const float* wv_w = (const float*)d_in[8];
  const float* wv_b = (const float*)d_in[9];
  const float* wo_w = (const float*)d_in[10];
  const float* wo_b = (const float*)d_in[11];

  float* out  = (float*)d_out;
  float* attn = out + (size_t)BL_ * D_;          // outputs: (out, attn) concatenated

  half_t* ws = (half_t*)d_ws;
  size_t off = 0;
  half_t* q16  = ws + off; off += (size_t)BL_ * D_;
  half_t* k16  = ws + off; off += (size_t)BL_ * D_;
  half_t* v16  = ws + off; off += (size_t)BL_ * D_;
  half_t* wq16 = ws + off; off += (size_t)D_ * D_;
  half_t* wk16 = ws + off; off += (size_t)D_ * D_;
  half_t* wv16 = ws + off; off += (size_t)D_ * D_;
  half_t* wo16 = ws + off; off += (size_t)D_ * D_;
  half_t* Qp   = ws + off; off += (size_t)BH_ * L_ * DH_;
  half_t* Kp   = ws + off; off += (size_t)BH_ * L_ * DH_;
  half_t* Vp   = ws + off; off += (size_t)BH_ * L_ * DH_;
  half_t* Oh   = ws + off; off += (size_t)BL_ * D_;

  auto cvt = [&](const float* s, half_t* d, int n) {
    const int n4 = n / 4;
    cvt_f32_f16<<<dim3((n4 + 255) / 256), 256, 0, stream>>>(s, d, n4);
  };
  cvt(q, q16, BL_ * D_);
  cvt(k, k16, BL_ * D_);
  cvt(v, v16, BL_ * D_);
  cvt(wq_w, wq16, D_ * D_);
  cvt(wk_w, wk16, D_ * D_);
  cvt(wv_w, wv16, D_ * D_);
  cvt(wo_w, wo16, D_ * D_);

  const dim3 gproj(D_ / TS, BL_ / TM);           // (8, 64)
  proj_gemm<<<gproj, 128, 0, stream>>>(q16, wq16, wq_b, Qp);
  proj_gemm<<<gproj, 128, 0, stream>>>(k16, wk16, wk_b, Kp);
  proj_gemm<<<gproj, 128, 0, stream>>>(v16, wv16, wv_b, Vp);

  const dim3 gsc(L_ / TS, L_ / TM, BH_);         // (32, 16, 32)
  scores_kernel<<<gsc, 128, 0, stream>>>(Qp, Kp, mask, attn);

  softmax_rows<<<dim3(BH_ * L_), 256, 0, stream>>>(attn);

  const dim3 gav(BH_, L_ / TM);                  // (32, 16)
  attnv_gemm<<<gav, 128, 0, stream>>>(attn, Vp, Oh);

  out_proj_gemm<<<gproj, 128, 0, stream>>>(Oh, wo16, wo_b, out);
}
